// Block_57604101374459
// MI455X (gfx1250) — compile-verified, matrix operand order checked by reading, not run on hardware
//
#include <hip/hip_runtime.h>
#include <stdint.h>

typedef _Float16 h16;
typedef __attribute__((ext_vector_type(8)))  _Float16 v8h;
typedef __attribute__((ext_vector_type(16))) _Float16 v16h;
typedef __attribute__((ext_vector_type(8)))  float    v8f;

#define N_EMBD 384
#define N_HEADS 6
#define HEAD 64
#define SEQ 256
#define BATCH 256
#define DFF 1536
#define NTOK (BATCH * SEQ)                 // 65536
static __device__ __constant__ float ATTN_SCALE = 19.595917942265423f; // sqrt(384)

// ---------- fragment helpers ----------
__device__ __forceinline__ v16h frag_ld(const h16* p) {
    union { v16h f; v8h h[2]; } u;
    u.h[0] = *(const v8h*)(p);
    u.h[1] = *(const v8h*)(p + 16);
    return u.f;
}

__device__ __forceinline__ v8f wmma_f16(v16h a, v16h b, v8f c) {
    return __builtin_amdgcn_wmma_f32_16x16x32_f16(false, a, false, b, (short)0, c, false, false);
}

// async 16B global -> LDS copy (ASYNCcnt-tracked, CDNA5 path)
__device__ __forceinline__ void async_ld16(uint32_t lds_off_bytes, const void* g) {
    asm volatile("global_load_async_to_lds_b128 %0, %1, off"
                 :: "v"(lds_off_bytes), "v"((uint64_t)(uintptr_t)g) : "memory");
}
__device__ __forceinline__ void wait_async0() {
    asm volatile("s_wait_asynccnt 0x0" ::: "memory");
}

// ---------- weight prep ----------
__global__ void prep_wqkv(const float* wq, const float* wk, const float* wv, h16* wt) {
    int idx = blockIdx.x * blockDim.x + threadIdx.x;     // over 1152*384
    if (idx >= 1152 * N_EMBD) return;
    int n = idx / N_EMBD, c = idx % N_EMBD;
    int sel = n / N_EMBD;                                 // 0=q 1=k 2=v
    int nn  = n % N_EMBD;
    int h = nn / HEAD, d = nn % HEAD;
    const float* src = (sel == 0) ? wq : (sel == 1) ? wk : wv;
    wt[idx] = (h16)src[(size_t)(h * N_EMBD + c) * HEAD + d];
}

__global__ void prep_t(const float* in, h16* out, int N, int K) {
    // in: [K, N] row-major (x @ W convention)  ->  out: [N, K] row-major (col-major B)
    int idx = blockIdx.x * blockDim.x + threadIdx.x;
    if (idx >= N * K) return;
    int n = idx / K, k = idx % K;
    out[idx] = (h16)in[(size_t)k * N + n];
}

// ---------- layernorm (wave per row) ----------
__global__ void ln_kernel(const float* __restrict__ x, const float* __restrict__ g,
                          const float* __restrict__ b, h16* __restrict__ out) {
    int wid = threadIdx.x >> 5, lane = threadIdx.x & 31;
    int row = blockIdx.x * 8 + wid;                       // blockDim = 256 -> 8 rows
    const float* xr = x + (size_t)row * N_EMBD;
    float v[12], s = 0.f, s2 = 0.f;
#pragma unroll
    for (int i = 0; i < 12; i++) { v[i] = xr[lane + 32 * i]; s += v[i]; s2 += v[i] * v[i]; }
#pragma unroll
    for (int o = 1; o < 32; o <<= 1) { s += __shfl_xor(s, o, 32); s2 += __shfl_xor(s2, o, 32); }
    float mean = s * (1.f / N_EMBD);
    float var  = s2 * (1.f / N_EMBD) - mean * mean;
    float rstd = rsqrtf(var + 1e-5f);
    h16* orow = out + (size_t)row * N_EMBD;
#pragma unroll
    for (int i = 0; i < 12; i++) {
        int c = lane + 32 * i;
        orow[c] = (h16)((v[i] - mean) * rstd * g[c] + b[c]);
    }
}

// ---------- WMMA GEMM core: async-staged, double-buffered LDS ----------
// Block tile 128(M) x 128(N), 8 waves as 4(M) x 2(N), wave tile 32x64 (2x4 WMMA tiles).
// LDS row stride 40 halves (80B = 20 banks) -> conflict-free ds_load_b128 fragments.
#define LDW   40
#define TILEH (128 * LDW)                 // halves per staged tile

__device__ __forceinline__ void gemm_core(const h16* __restrict__ A, const h16* __restrict__ Bt,
                                          int K, h16* sm, v8f (&acc)[2][4]) {
    int tid  = threadIdx.x;
    int lane = tid & 31, wid = tid >> 5;
    int wm = wid & 3, wn = wid >> 2;
    int mblk = blockIdx.x * 128, nblk = blockIdx.y * 128;
    int hl = lane & 15, ch = (lane >> 4) * 8;

    // staging: 512 x 16B chunks per tile; thread handles chunks tid and tid+256
    auto stage = [&](int kt, int buf) {
        int k0 = kt * 32;
        h16* base = sm + buf * 2 * TILEH;
#pragma unroll
        for (int j = 0; j < 2; j++) {
            int c   = tid + j * 256;
            int row = c >> 2, c8 = (c & 3) * 8;
            async_ld16((uint32_t)(uintptr_t)&base[row * LDW + c8],
                       A + (size_t)(mblk + row) * K + k0 + c8);
            async_ld16((uint32_t)(uintptr_t)&base[TILEH + row * LDW + c8],
                       Bt + (size_t)(nblk + row) * K + k0 + c8);
        }
    };

    stage(0, 0);
    int KT = K / 32;
    for (int kt = 0; kt < KT; kt++) {
        int buf = kt & 1;
        wait_async0();           // stage kt complete (this wave's copies)
        __syncthreads();         // all waves' staged data visible
        if (kt + 1 < KT) stage(kt + 1, buf ^ 1);   // overlap copy with compute

        const h16* sA = sm + buf * 2 * TILEH;
        const h16* sB = sA + TILEH;
        v16h af[2], bf[4];
#pragma unroll
        for (int i = 0; i < 2; i++)
            af[i] = frag_ld(sA + (wm * 32 + i * 16 + hl) * LDW + ch);
#pragma unroll
        for (int j = 0; j < 4; j++)
            bf[j] = frag_ld(sB + (wn * 64 + j * 16 + hl) * LDW + ch);
#pragma unroll
        for (int i = 0; i < 2; i++)
#pragma unroll
            for (int j = 0; j < 4; j++)
                acc[i][j] = wmma_f16(af[i], bf[j], acc[i][j]);
        __syncthreads();         // done reading buf before it is restaged
    }
}

// MODE 1: out f16 = relu(acc + bias)      (FFN layer 1)
// MODE 2: out f32 = acc + bias + resid    (proj / FFN layer 2 with residual)
template <int MODE>
__global__ void gemm_kernel(const h16* __restrict__ A, const h16* __restrict__ Bt,
                            const float* __restrict__ bias, const float* __restrict__ resid,
                            h16* __restrict__ outH, float* __restrict__ outF, int K, int N) {
    __shared__ h16 sm[2 * 2 * TILEH];
    v8f acc[2][4] = {};
    gemm_core(A, Bt, K, sm, acc);

    int lane = threadIdx.x & 31, wid = threadIdx.x >> 5;
    int wm = wid & 3, wn = wid >> 2;
    int hl = lane & 15, hi = lane >> 4;
    int mrow = blockIdx.x * 128 + wm * 32 + 8 * hi;
    int ncol = blockIdx.y * 128 + wn * 64 + hl;
#pragma unroll
    for (int tm = 0; tm < 2; tm++) {
#pragma unroll
        for (int tn = 0; tn < 4; tn++) {
            int m0 = mrow + tm * 16;
            int n  = ncol + tn * 16;
            float bv = bias[n];
#pragma unroll
            for (int r = 0; r < 8; r++) {
                size_t off = (size_t)(m0 + r) * N + n;
                float v = acc[tm][tn][r] + bv;
                if (MODE == 1) {
                    outH[off] = (h16)(v > 0.f ? v : 0.f);
                } else {
                    outF[off] = v + resid[off];
                }
            }
        }
    }
}

// ---------- QKV GEMM with scatter epilogue ----------
// A = h16 [NTOK, 384]; Bt = wqkvT [1152, 384]. n<384 -> q, <768 -> k, else v.
// q,k stored [B,H,T,64]; v stored transposed [B,H,64,T].
__global__ void gemm_qkv(const h16* __restrict__ A, const h16* __restrict__ Bt,
                         h16* __restrict__ q, h16* __restrict__ k, h16* __restrict__ vt) {
    __shared__ h16 sm[2 * 2 * TILEH];
    v8f acc[2][4] = {};
    gemm_core(A, Bt, N_EMBD, sm, acc);

    int lane = threadIdx.x & 31, wid = threadIdx.x >> 5;
    int wm = wid & 3, wn = wid >> 2;
    int hl = lane & 15, hi = lane >> 4;
    int mrow = blockIdx.x * 128 + wm * 32 + 8 * hi;
    int ncol = blockIdx.y * 128 + wn * 64 + hl;
#pragma unroll
    for (int tm = 0; tm < 2; tm++) {
#pragma unroll
        for (int tn = 0; tn < 4; tn++) {
            int m0 = mrow + tm * 16;
            int n  = ncol + tn * 16;
            int sel = n / N_EMBD, nn = n % N_EMBD;
            int hh = nn / HEAD, dd = nn % HEAD;
#pragma unroll
            for (int r = 0; r < 8; r++) {
                int m = m0 + r;
                int bI = m >> 8, t = m & 255;                 // T = 256
                h16 val = (h16)acc[tm][tn][r];
                if (sel == 0)
                    q[((size_t)((bI * N_HEADS + hh) * SEQ + t)) * HEAD + dd] = val;
                else if (sel == 1)
                    k[((size_t)((bI * N_HEADS + hh) * SEQ + t)) * HEAD + dd] = val;
                else
                    vt[((size_t)((bI * N_HEADS + hh) * HEAD + dd)) * SEQ + t] = val;
            }
        }
    }
}

// ---------- causal flash attention ----------
// grid = B*H*4 blocks, 128 threads (4 waves); wave owns 16 q-rows.
__global__ void attn_kernel(const h16* __restrict__ q, const h16* __restrict__ k,
                            const h16* __restrict__ vt, h16* __restrict__ att) {
    __shared__ h16 lds[4 * 16 * 32];
    int wid = threadIdx.x >> 5, lane = threadIdx.x & 31;
    int bh = blockIdx.x >> 2;
    int qt = blockIdx.x & 3;
    int qb = qt * 64 + wid * 16;
    int hl = lane & 15, hi = lane >> 4, ch = hi * 8;

    const h16* qbase = q  + (size_t)bh * SEQ * HEAD;
    const h16* kbase = k  + (size_t)bh * SEQ * HEAD;
    const h16* vbase = vt + (size_t)bh * HEAD * SEQ;

    const h16* qrow = qbase + (size_t)(qb + hl) * HEAD + ch;
    v16h qf0 = frag_ld(qrow);
    v16h qf1 = frag_ld(qrow + 32);

    float rm[8], rl[8];
    v8f o0 = {}, o1 = {}, o2 = {}, o3 = {};
#pragma unroll
    for (int r = 0; r < 8; r++) { rm[r] = -1e30f; rl[r] = 0.f; }

    int qrow0 = qb + 8 * hi;
    int s_end = qb + 16;
    int ldsbase = wid * 16 * 32;

    for (int s0 = 0; s0 < s_end; s0 += 32) {
        const h16* k0p = kbase + (size_t)(s0 + hl) * HEAD + ch;
        const h16* k1p = k0p + 16 * HEAD;
        v8f sA = {}, sB = {};
        sA = wmma_f16(qf0, frag_ld(k0p),      sA);
        sA = wmma_f16(qf1, frag_ld(k0p + 32), sA);
        sB = wmma_f16(qf0, frag_ld(k1p),      sB);
        sB = wmma_f16(qf1, frag_ld(k1p + 32), sB);

        int key0 = s0 + hl, key1 = key0 + 16;
        float mx[8];
#pragma unroll
        for (int r = 0; r < 8; r++) {
            float a = sA[r] * ATTN_SCALE, b = sB[r] * ATTN_SCALE;
            int qg = qrow0 + r;
            if (key0 > qg) a = -1e30f;
            if (key1 > qg) b = -1e30f;
            sA[r] = a; sB[r] = b;
            mx[r] = fmaxf(a, b);
        }
#pragma unroll
        for (int off = 1; off < 16; off <<= 1)
#pragma unroll
            for (int r = 0; r < 8; r++) mx[r] = fmaxf(mx[r], __shfl_xor(mx[r], off, 32));

        float alpha[8], rs[8];
#pragma unroll
        for (int r = 0; r < 8; r++) {
            float mn = fmaxf(rm[r], mx[r]);
            alpha[r] = __expf(rm[r] - mn);
            rm[r] = mn;
            float a = __expf(sA[r] - mn);
            float b = __expf(sB[r] - mn);
            sA[r] = a; sB[r] = b;
            rs[r] = a + b;
        }
#pragma unroll
        for (int off = 1; off < 16; off <<= 1)
#pragma unroll
            for (int r = 0; r < 8; r++) rs[r] += __shfl_xor(rs[r], off, 32);
#pragma unroll
        for (int r = 0; r < 8; r++) {
            rl[r] = rl[r] * alpha[r] + rs[r];
            o0[r] *= alpha[r]; o1[r] *= alpha[r]; o2[r] *= alpha[r]; o3[r] *= alpha[r];
        }

        // P (D-layout) -> LDS -> A-layout fragment
#pragma unroll
        for (int r = 0; r < 8; r++) {
            int m = r + 8 * hi;
            lds[ldsbase + m * 32 + hl]      = (h16)sA[r];
            lds[ldsbase + m * 32 + 16 + hl] = (h16)sB[r];
        }
        asm volatile("s_wait_dscnt 0" ::: "memory");
        union { v16h f; v8h h[2]; } up;
        up.h[0] = *(const v8h*)&lds[ldsbase + hl * 32 + ch];
        up.h[1] = *(const v8h*)&lds[ldsbase + hl * 32 + ch + 16];
        v16h pf = up.f;
        asm volatile("" ::: "memory");

        const h16* vrow = vbase + (size_t)hl * SEQ + s0 + ch;
        o0 = wmma_f16(pf, frag_ld(vrow),            o0);
        o1 = wmma_f16(pf, frag_ld(vrow + 16 * SEQ), o1);
        o2 = wmma_f16(pf, frag_ld(vrow + 32 * SEQ), o2);
        o3 = wmma_f16(pf, frag_ld(vrow + 48 * SEQ), o3);
    }

    int b = bh / N_HEADS, h = bh % N_HEADS;
#pragma unroll
    for (int r = 0; r < 8; r++) {
        float inv = 1.0f / rl[r];
        int t = qb + 8 * hi + r;
        h16* orow = att + ((size_t)(b * SEQ + t)) * N_EMBD + h * HEAD;
        orow[hl]      = (h16)(o0[r] * inv);
        orow[16 + hl] = (h16)(o1[r] * inv);
        orow[32 + hl] = (h16)(o2[r] * inv);
        orow[48 + hl] = (h16)(o3[r] * inv);
    }
}

// ---------- launch ----------
extern "C" void kernel_launch(void* const* d_in, const int* in_sizes, int n_in,
                              void* d_out, int out_size, void* d_ws, size_t ws_size,
                              hipStream_t stream) {
    (void)in_sizes; (void)n_in; (void)out_size; (void)ws_size;
    const float* x      = (const float*)d_in[0];
    const float* ln1_g  = (const float*)d_in[1];
    const float* ln1_b  = (const float*)d_in[2];
    const float* wq     = (const float*)d_in[3];
    const float* wk     = (const float*)d_in[4];
    const float* wv     = (const float*)d_in[5];
    const float* w_proj = (const float*)d_in[6];
    const float* b_proj = (const float*)d_in[7];
    const float* ln2_g  = (const float*)d_in[8];
    const float* ln2_b  = (const float*)d_in[9];
    const float* w1     = (const float*)d_in[10];
    const float* b1     = (const float*)d_in[11];
    const float* w2     = (const float*)d_in[12];
    const float* b2     = (const float*)d_in[13];
    float* out = (float*)d_out;

    uint8_t* ws = (uint8_t*)d_ws;
    const size_t SZ_H16 = (size_t)NTOK * N_EMBD * 2;       // 48 MB
    h16*   hbuf  = (h16*)(ws);                             // h / h2 (reused)
    h16*   qb    = (h16*)(ws + SZ_H16);                    // q
    h16*   kb    = (h16*)(ws + 2 * SZ_H16);                // k
    h16*   vtb   = (h16*)(ws + 3 * SZ_H16);                // v^T
    h16*   attb  = (h16*)(ws + 4 * SZ_H16);                // attention out
    float* x1    = (float*)(ws + 5 * SZ_H16);              // 96 MB f32 residual
    uint8_t* wsp = ws + 5 * SZ_H16 + (size_t)NTOK * N_EMBD * 4;
    h16* wqkvT  = (h16*)(wsp);                              wsp += (size_t)1152 * N_EMBD * 2;
    h16* wprojT = (h16*)(wsp);                              wsp += (size_t)N_EMBD * N_EMBD * 2;
    h16* w1T    = (h16*)(wsp);                              wsp += (size_t)DFF * N_EMBD * 2;
    h16* w2T    = (h16*)(wsp);
    // ff16 aliases the dead q/k/vt/att region (192 MB) — stream order makes this safe
    h16* ffb = qb;

    // weight prep
    prep_wqkv<<<(1152 * N_EMBD) / 256, 256, 0, stream>>>(wq, wk, wv, wqkvT);
    prep_t<<<(N_EMBD * N_EMBD) / 256, 256, 0, stream>>>(w_proj, wprojT, N_EMBD, N_EMBD);
    prep_t<<<(DFF * N_EMBD) / 256, 256, 0, stream>>>(w1, w1T, DFF, N_EMBD);
    prep_t<<<(N_EMBD * DFF) / 256, 256, 0, stream>>>(w2, w2T, N_EMBD, DFF);

    // LN1 -> h (f16)
    ln_kernel<<<NTOK / 8, 256, 0, stream>>>(x, ln1_g, ln1_b, hbuf);

    // QKV projection (block tile 128x128)
    gemm_qkv<<<dim3(NTOK / 128, 1152 / 128), 256, 0, stream>>>(hbuf, wqkvT, qb, kb, vtb);

    // causal attention
    attn_kernel<<<BATCH * N_HEADS * 4, 128, 0, stream>>>(qb, kb, vtb, attb);

    // out projection + residual -> x1 (f32)
    gemm_kernel<2><<<dim3(NTOK / 128, N_EMBD / 128), 256, 0, stream>>>(
        attb, wprojT, b_proj, x, nullptr, x1, N_EMBD, N_EMBD);

    // LN2 -> h2 (f16, reuse hbuf)
    ln_kernel<<<NTOK / 8, 256, 0, stream>>>(x1, ln2_g, ln2_b, hbuf);

    // FFN layer 1: relu(h2 @ w1 + b1) -> ff (f16)
    gemm_kernel<1><<<dim3(NTOK / 128, DFF / 128), 256, 0, stream>>>(
        hbuf, w1T, b1, nullptr, ffb, nullptr, N_EMBD, DFF);

    // FFN layer 2 + residual -> out (f32)
    gemm_kernel<2><<<dim3(NTOK / 128, N_EMBD / 128), 256, 0, stream>>>(
        ffb, w2T, b2, x1, nullptr, out, DFF, N_EMBD);
}
